// sLSTMCell_27848567947721
// MI455X (gfx1250) — matrix operand order repaired and checked by reading, not compiled, not run on hardware
//
#include <hip/hip_runtime.h>
#include <hip/hip_bf16.h>
#include <math.h>
#include <stdint.h>

typedef __attribute__((ext_vector_type(16))) _Float16 v16h;
typedef __attribute__((ext_vector_type(8)))  _Float16 v8h;
typedef __attribute__((ext_vector_type(8)))  float    v8f;
typedef __attribute__((ext_vector_type(4)))  unsigned int u32x4;
typedef __attribute__((ext_vector_type(8)))  int      i32x8;
typedef __attribute__((ext_vector_type(4)))  int      i32x4;

#define BB   8192
#define DD   1024
#define HH   8
#define DH   128
#define UU   1365
#define UP   1408          // 22 * 64, padded U
#define BD   8388608UL     // BB*DD = 2^23

// ---------------- WMMA fragment helpers (wave32, 16x16x32 f16) ----------------

__device__ __forceinline__ v16h load_a_frag(const _Float16* __restrict__ A, int lda,
                                            int row, int k0, int lane) {
  // A-matrix 16x32 f16 layout: lanes 0-15 -> M=lane, K = k0+{0..7, 16..23}
  //                            lanes 16-31 -> M=lane&15, K = k0+{8..15, 24..31}
  int kk = k0 + ((lane >> 4) << 3);
  const _Float16* p = A + (size_t)row * lda + kk;
  union { v16h v; v8h h[2]; } u;
  u.h[0] = *(const v8h*)(p);
  u.h[1] = *(const v8h*)(p + 16);
  return u.v;
}

__device__ __forceinline__ v16h load_b_frag(const _Float16* __restrict__ Bm, int ldb,
                                            int k0, int nb, int lane) {
  // B-matrix 32x16 f16 layout: lanes 0-15 -> N=nb+lane, K = k0+{0..7, 16..23}
  //                            lanes 16-31 -> N=nb+(lane&15), K = k0+{8..15, 24..31}
  const int n = nb + (lane & 15);
  const int kb = k0 + ((lane >> 4) << 3);
  v16h b;
#pragma unroll
  for (int s = 0; s < 8; ++s) {
    b[s] = Bm[(size_t)(kb + s) * ldb + n];
    b[s + 8] = Bm[(size_t)(kb + 16 + s) * ldb + n];
  }
  return b;
}

__device__ __forceinline__ v8f wmma_f16(v16h a, v16h b, v8f c) {
  return __builtin_amdgcn_wmma_f32_16x16x32_f16(false, a, false, b, (short)0, c,
                                                false, false);
}

// ---------------- Tensor Data Mover: async 2-D tile load Global -> LDS --------
// Builds a D# (groups 0/1; 2-D tile so groups 2/3 are zero) and issues
// TENSOR_LOAD_TO_LDS. data_size=2 bytes. Tracked by TENSORcnt.

__device__ __forceinline__ void tdm_load_tile_f16(const _Float16* gptr,
                                                  unsigned int lds_off,
                                                  unsigned int tdim0,
                                                  unsigned int tdim1,
                                                  unsigned int tile0,
                                                  unsigned int tile1,
                                                  unsigned int stride0) {
  uint64_t ga = (uint64_t)(uintptr_t)gptr;
  u32x4 g0;
  g0[0] = 1u;                                          // count=1, user descriptor
  g0[1] = lds_off;                                     // lds_addr [63:32]
  g0[2] = (unsigned int)ga;                            // global_addr lo
  g0[3] = (unsigned int)((ga >> 32) & 0x01FFFFFFu) | (2u << 30);  // hi + type=2
  i32x8 g1;
  g1[0] = (int)(1u << 16);                             // data_size=1 (2 bytes)
  g1[1] = (int)((tdim0 & 0xFFFFu) << 16);              // tensor_dim0[15:0] @ [63:48]
  g1[2] = (int)((tdim0 >> 16) | ((tdim1 & 0xFFFFu) << 16));
  g1[3] = (int)((tdim1 >> 16) | (tile0 << 16));        // tile_dim0 @ [127:112]
  g1[4] = (int)(tile1);                                // tile_dim1; tile_dim2=0
  g1[5] = (int)(stride0);                              // tensor_dim0_stride[31:0]
  g1[6] = 0;                                           // stride0[47:32], stride1 lo
  g1[7] = 0;
  i32x4 z4 = {0, 0, 0, 0};
#if __clang_major__ >= 23
  i32x8 z8 = {0, 0, 0, 0, 0, 0, 0, 0};
  __builtin_amdgcn_tensor_load_to_lds(g0, g1, z4, z4, z8, 0);
#else
  __builtin_amdgcn_tensor_load_to_lds(g0, g1, z4, z4, 0);
#endif
}

__device__ __forceinline__ unsigned int lds_offset_of(const void* p) {
  // low 32 bits of a flat LDS pointer are the offset within the WG allocation
  return (unsigned int)(uintptr_t)p;
}

// ---------------- Kernel 1: LayerNorm + causal conv1d(4) + SiLU ----------------

__global__ __launch_bounds__(128) void ln_conv_kernel(
    const float* __restrict__ x, const float* __restrict__ g,
    const float* __restrict__ bta, const float* __restrict__ cw,
    const float* __restrict__ cb, _Float16* __restrict__ x_h,
    _Float16* __restrict__ xc_h) {
  __shared__ float s_xn[DD];
  __shared__ float s_r0[128];
  __shared__ float s_r1[128];
  __shared__ float s_mu, s_rs;
  const int t = threadIdx.x;
  const int b = blockIdx.x;
  const float* xr = x + (size_t)b * DD;
  float xv[8];
  float sum = 0.f, sq = 0.f;
#pragma unroll
  for (int q = 0; q < 8; ++q) {
    int j = t + 128 * q;
    float v = xr[j];
    xv[q] = v; sum += v; sq += v * v;
  }
  s_r0[t] = sum; s_r1[t] = sq;
  __syncthreads();
  for (int off = 64; off > 0; off >>= 1) {
    if (t < off) { s_r0[t] += s_r0[t + off]; s_r1[t] += s_r1[t + off]; }
    __syncthreads();
  }
  if (t == 0) {
    float mu = s_r0[0] * (1.f / DD);
    float var = s_r1[0] * (1.f / DD) - mu * mu;
    s_mu = mu; s_rs = rsqrtf(var + 1e-5f);
  }
  __syncthreads();
  float mu = s_mu, rs = s_rs;
#pragma unroll
  for (int q = 0; q < 8; ++q) {
    int j = t + 128 * q;
    float xn = (xv[q] - mu) * rs * g[j] + bta[j];
    s_xn[j] = xn;
    x_h[(size_t)b * DD + j] = (_Float16)xv[q];
  }
  __syncthreads();
  float w0 = cw[0], w1 = cw[1], w2 = cw[2], w3 = cw[3], c0 = cb[0];
#pragma unroll
  for (int q = 0; q < 8; ++q) {
    int j = t + 128 * q;
    float a = c0 + w3 * s_xn[j];
    if (j >= 1) a += w2 * s_xn[j - 1];
    if (j >= 2) a += w1 * s_xn[j - 2];
    if (j >= 3) a += w0 * s_xn[j - 3];
    float sv = a / (1.f + expf(-a));  // silu
    xc_h[(size_t)b * DD + j] = (_Float16)sv;
  }
}

// ---------------- Converters ----------------

__global__ void cvt_f16_kernel(const float* __restrict__ s, _Float16* __restrict__ d,
                               size_t n) {
  size_t i = (size_t)blockIdx.x * blockDim.x + threadIdx.x;
  if (i < n) d[i] = (_Float16)s[i];
}

__global__ void pad_cvt_kernel(const float* __restrict__ s, _Float16* __restrict__ d,
                               int srows, int scols, int drows, int dcols) {
  size_t i = (size_t)blockIdx.x * blockDim.x + threadIdx.x;
  size_t total = (size_t)drows * dcols;
  if (i >= total) return;
  int r = (int)(i / dcols);
  int c = (int)(i % dcols);
  _Float16 v = (_Float16)0.f;
  if (r < srows && c < scols) v = (_Float16)s[(size_t)r * scols + c];
  d[i] = v;
}

// ---------------- Kernel 2: block-diagonal gate GEMMs ----------------
// pre[g] = A_g @ W[g,h] + h_prev @ R[g,h] + (bW[g]+bR[g]);  A_{z,o}=x, A_{i,f}=x_conv
// grid: (B/64, 2, 32=gate*8+head), block 128 (4 waves); wave tile 16x64.
// 128x128 weights are fully L2/L0 resident -> direct loads.

__global__ __launch_bounds__(128) void gate_gemm_kernel(
    const _Float16* __restrict__ x_h, const _Float16* __restrict__ xc_h,
    const _Float16* __restrict__ hp_h, const _Float16* __restrict__ W16,
    const _Float16* __restrict__ R16, const float* __restrict__ bW,
    const float* __restrict__ bR, float* __restrict__ gates) {
  const int lane = threadIdx.x & 31;
  const int wave = threadIdx.x >> 5;
  const int g = blockIdx.z >> 3, h = blockIdx.z & 7;
  const int mBase = blockIdx.x * 64 + wave * 16;
  const int nb0 = blockIdx.y * 64;
  const int acol = h * DH;
  const _Float16* A1 = ((g == 0) || (g == 3)) ? x_h : xc_h;
  const _Float16* Wg = W16 + ((size_t)(g * HH + h) << 14);  // 128*128
  const _Float16* Rg = R16 + ((size_t)(g * HH + h) << 14);
  const int m = mBase + (lane & 15);

  v8f acc[4];
#pragma unroll
  for (int n = 0; n < 4; ++n) acc[n] = v8f{0.f, 0.f, 0.f, 0.f, 0.f, 0.f, 0.f, 0.f};

#pragma unroll
  for (int k0 = 0; k0 < DH; k0 += 32) {
    v16h a = load_a_frag(A1 + acol, DD, m, k0, lane);
#pragma unroll
    for (int n = 0; n < 4; ++n)
      acc[n] = wmma_f16(a, load_b_frag(Wg, DH, k0, nb0 + 16 * n, lane), acc[n]);
  }
#pragma unroll
  for (int k0 = 0; k0 < DH; k0 += 32) {
    v16h a = load_a_frag(hp_h + acol, DD, m, k0, lane);
#pragma unroll
    for (int n = 0; n < 4; ++n)
      acc[n] = wmma_f16(a, load_b_frag(Rg, DH, k0, nb0 + 16 * n, lane), acc[n]);
  }

#pragma unroll
  for (int n = 0; n < 4; ++n) {
    int jc = acol + nb0 + 16 * n + (lane & 15);
    float bias = bW[g * DD + jc] + bR[g * DD + jc];
#pragma unroll
    for (int r = 0; r < 8; ++r) {
      int row = mBase + r + ((lane >> 4) << 3);
      gates[((size_t)g << 23) + (size_t)row * DD + jc] = acc[n][r] + bias;
    }
  }
}

// ---------------- Kernel 3: sLSTM state update (elementwise) ----------------

__global__ void gate_elem_kernel(const float* __restrict__ gates,
                                const float* __restrict__ c_prev,
                                const float* __restrict__ n_prev,
                                const float* __restrict__ m_prev,
                                float* __restrict__ out,
                                _Float16* __restrict__ ht_h) {
  size_t idx = (size_t)blockIdx.x * blockDim.x + threadIdx.x;
  if (idx >= BD) return;
  float zp = gates[idx];
  float ip = gates[BD + idx];
  float fp = gates[2 * BD + idx];
  float op = gates[3 * BD + idx];
  float mp = m_prev[idx];
  float z = tanhf(zp);
  float o = 1.f / (1.f + expf(-op));
  float mt = fmaxf(fp + mp, ip);
  float i = expf(ip - mt);
  float f = expf(fp + mp - mt);
  float ct = f * c_prev[idx] + i * z;
  float nt = f * n_prev[idx] + i;
  float ht = o * ct / nt;
  out[BD + idx] = ht;
  out[2 * BD + idx] = ct;
  out[3 * BD + idx] = nt;
  out[4 * BD + idx] = mt;
  ht_h[idx] = (_Float16)ht;
}

// ---------------- Kernel 4: fused up-projections + gelu gate ----------------
// left = h@upl + bl ; right = gelu(h@upr + br) ; prod = left*right (f16, padded N)
// B tiles (32K x 64N, both matrices) staged via TDM into LDS, double-buffered.

__global__ __launch_bounds__(128) void up_gemm_kernel(
    const _Float16* __restrict__ ht, const _Float16* __restrict__ upl,
    const _Float16* __restrict__ upr, const float* __restrict__ bl,
    const float* __restrict__ br, _Float16* __restrict__ prod) {
  __shared__ _Float16 sB[2][2][32 * 64];  // [buf][mat L/R][k*64+n]
  const int lane = threadIdx.x & 31;
  const int wave = threadIdx.x >> 5;
  const int mBase = blockIdx.x * 64 + wave * 16;
  const int nb0 = blockIdx.y * 64;
  const int m = mBase + (lane & 15);
  const int nK = DD / 32;

  v8f accL[4], accR[4];
#pragma unroll
  for (int n = 0; n < 4; ++n) {
    accL[n] = v8f{0.f, 0.f, 0.f, 0.f, 0.f, 0.f, 0.f, 0.f};
    accR[n] = v8f{0.f, 0.f, 0.f, 0.f, 0.f, 0.f, 0.f, 0.f};
  }

  if (wave == 0) {
    tdm_load_tile_f16(upl + nb0, lds_offset_of(&sB[0][0][0]), UP, DD, 64, 32, UP);
    tdm_load_tile_f16(upr + nb0, lds_offset_of(&sB[0][1][0]), UP, DD, 64, 32, UP);
  }

  for (int i = 0; i < nK; ++i) {
    if (wave == 0) __builtin_amdgcn_s_wait_tensorcnt(0);
    __syncthreads();  // buf[i&1] ready for all waves
    if (wave == 0 && (i + 1) < nK) {
      int kn = (i + 1) * 32;
      tdm_load_tile_f16(upl + (size_t)kn * UP + nb0,
                        lds_offset_of(&sB[(i + 1) & 1][0][0]), UP, DD, 64, 32, UP);
      tdm_load_tile_f16(upr + (size_t)kn * UP + nb0,
                        lds_offset_of(&sB[(i + 1) & 1][1][0]), UP, DD, 64, 32, UP);
    }
    const int buf = i & 1;
    v16h a = load_a_frag(ht, DD, m, i * 32, lane);
    const _Float16* pl = &sB[buf][0][0];
    const _Float16* pr = &sB[buf][1][0];
#pragma unroll
    for (int n = 0; n < 4; ++n) {
      v16h bL = load_b_frag(pl, 64, 0, 16 * n, lane);
      accL[n] = wmma_f16(a, bL, accL[n]);
      v16h bR = load_b_frag(pr, 64, 0, 16 * n, lane);
      accR[n] = wmma_f16(a, bR, accR[n]);
    }
    __syncthreads();  // done reading buf before it is refilled at i+2
  }

#pragma unroll
  for (int n = 0; n < 4; ++n) {
    int j = nb0 + 16 * n + (lane & 15);
    float bll = (j < UU) ? bl[j] : 0.f;
    float brr = (j < UU) ? br[j] : 0.f;
#pragma unroll
    for (int r = 0; r < 8; ++r) {
      int row = mBase + r + ((lane >> 4) << 3);
      float left = accL[n][r] + bll;
      float rv = accR[n][r] + brr;
      float right = 0.5f * rv * (1.f + erff(rv * 0.70710678118654752f));  // exact gelu
      prod[(size_t)row * UP + j] = (_Float16)(left * right);
    }
  }
}

// ---------------- Kernel 5: down projection (TDM-staged B) ----------------

__global__ __launch_bounds__(128) void down_gemm_kernel(
    const _Float16* __restrict__ prod, const _Float16* __restrict__ dw,
    const float* __restrict__ db, float* __restrict__ out) {
  __shared__ _Float16 sB[2][32 * 64];
  const int lane = threadIdx.x & 31;
  const int wave = threadIdx.x >> 5;
  const int mBase = blockIdx.x * 64 + wave * 16;
  const int nb0 = blockIdx.y * 64;
  const int m = mBase + (lane & 15);
  const int nK = UP / 32;  // 44

  v8f acc[4];
#pragma unroll
  for (int n = 0; n < 4; ++n) acc[n] = v8f{0.f, 0.f, 0.f, 0.f, 0.f, 0.f, 0.f, 0.f};

  if (wave == 0)
    tdm_load_tile_f16(dw + nb0, lds_offset_of(&sB[0][0]), DD, UP, 64, 32, DD);

  for (int i = 0; i < nK; ++i) {
    if (wave == 0) __builtin_amdgcn_s_wait_tensorcnt(0);
    __syncthreads();
    if (wave == 0 && (i + 1) < nK) {
      int kn = (i + 1) * 32;
      tdm_load_tile_f16(dw + (size_t)kn * DD + nb0,
                        lds_offset_of(&sB[(i + 1) & 1][0]), DD, UP, 64, 32, DD);
    }
    const int buf = i & 1;
    v16h a = load_a_frag(prod, UP, m, i * 32, lane);
    const _Float16* pb = &sB[buf][0];
#pragma unroll
    for (int n = 0; n < 4; ++n) {
      v16h bf = load_b_frag(pb, 64, 0, 16 * n, lane);
      acc[n] = wmma_f16(a, bf, acc[n]);
    }
    __syncthreads();
  }

#pragma unroll
  for (int n = 0; n < 4; ++n) {
    int j = nb0 + 16 * n + (lane & 15);
    float bias = db[j];
#pragma unroll
    for (int r = 0; r < 8; ++r) {
      int row = mBase + r + ((lane >> 4) << 3);
      out[(size_t)row * DD + j] = acc[n][r] + bias;
    }
  }
}

// ---------------- launch ----------------

extern "C" void kernel_launch(void* const* d_in, const int* in_sizes, int n_in,
                              void* d_out, int out_size, void* d_ws, size_t ws_size,
                              hipStream_t stream) {
  const float* x      = (const float*)d_in[0];
  const float* h_prev = (const float*)d_in[1];
  const float* c_prev = (const float*)d_in[2];
  const float* n_prev = (const float*)d_in[3];
  const float* m_prev = (const float*)d_in[4];
  const float* ln_g   = (const float*)d_in[5];
  const float* ln_b   = (const float*)d_in[6];
  const float* conv_w = (const float*)d_in[7];
  const float* conv_b = (const float*)d_in[8];
  const float* W      = (const float*)d_in[9];
  const float* bW     = (const float*)d_in[10];
  const float* R      = (const float*)d_in[11];
  const float* bR     = (const float*)d_in[12];
  const float* upl_w  = (const float*)d_in[13];
  const float* upl_b  = (const float*)d_in[14];
  const float* upr_w  = (const float*)d_in[15];
  const float* upr_b  = (const float*)d_in[16];
  const float* down_w = (const float*)d_in[17];
  const float* down_b = (const float*)d_in[18];
  float* out = (float*)d_out;

  // workspace carve-up (256B aligned)
  char* base = (char*)d_ws;
  size_t off = 0;
  auto take = [&](size_t bytes) {
    size_t cur = off;
    off += (bytes + 255) & ~(size_t)255;
    return (void*)(base + cur);
  };
  const size_t WRN = 4UL * HH * DH * DH;  // 524288
  _Float16* x_h    = (_Float16*)take(BD * 2);
  _Float16* xc_h   = (_Float16*)take(BD * 2);
  _Float16* hp_h   = (_Float16*)take(BD * 2);
  _Float16* ht_h   = (_Float16*)take(BD * 2);
  _Float16* W16    = (_Float16*)take(WRN * 2);
  _Float16* R16    = (_Float16*)take(WRN * 2);
  _Float16* upl16  = (_Float16*)take((size_t)DD * UP * 2);
  _Float16* upr16  = (_Float16*)take((size_t)DD * UP * 2);
  _Float16* down16 = (_Float16*)take((size_t)UP * DD * 2);
  float*    gates  = (float*)take(4UL * BD * 4);
  _Float16* prod   = (_Float16*)take((size_t)BB * UP * 2);
  (void)ws_size; (void)n_in; (void)in_sizes; (void)out_size;

  // 1) LayerNorm + conv + silu (+ f16 copy of x)
  ln_conv_kernel<<<BB, 128, 0, stream>>>(x, ln_g, ln_b, conv_w, conv_b, x_h, xc_h);

  // 2) f16 conversions / padded weight conversions
  cvt_f16_kernel<<<(BD + 255) / 256, 256, 0, stream>>>(h_prev, hp_h, BD);
  cvt_f16_kernel<<<(WRN + 255) / 256, 256, 0, stream>>>(W, W16, WRN);
  cvt_f16_kernel<<<(WRN + 255) / 256, 256, 0, stream>>>(R, R16, WRN);
  {
    size_t nu = (size_t)DD * UP;
    pad_cvt_kernel<<<(nu + 255) / 256, 256, 0, stream>>>(upl_w, upl16, DD, UU, DD, UP);
    pad_cvt_kernel<<<(nu + 255) / 256, 256, 0, stream>>>(upr_w, upr16, DD, UU, DD, UP);
    size_t nd = (size_t)UP * DD;
    pad_cvt_kernel<<<(nd + 255) / 256, 256, 0, stream>>>(down_w, down16, UU, DD, UP, DD);
  }

  // 3) block-diagonal gate GEMMs (4 gates x 8 heads)
  gate_gemm_kernel<<<dim3(BB / 64, DH / 64, 4 * HH), 128, 0, stream>>>(
      x_h, xc_h, hp_h, W16, R16, bW, bR, gates);

  // 4) sLSTM state update
  gate_elem_kernel<<<(BD + 255) / 256, 256, 0, stream>>>(gates, c_prev, n_prev,
                                                         m_prev, out, ht_h);

  // 5) fused up-projections + gelu gating (TDM-staged weights)
  up_gemm_kernel<<<dim3(BB / 64, UP / 64), 128, 0, stream>>>(ht_h, upl16, upr16,
                                                             upl_b, upr_b, prod);

  // 6) down projection -> output segment 0 (TDM-staged weights)
  down_gemm_kernel<<<dim3(BB / 64, DD / 64), 128, 0, stream>>>(prod, down16,
                                                               down_b, out);
}
